// SchrodingerAttention_85882166050816
// MI455X (gfx1250) — matrix operand order repaired and checked by reading, not compiled, not run on hardware
//
#include <hip/hip_runtime.h>
#include <math.h>

// ---------------------------------------------------------------------------
// Problem constants (fixed by the reference): B=8, L=2048, D=512, NB=4, MP=8
// ---------------------------------------------------------------------------
#define B_   8
#define L_   2048
#define D_   512
#define M_   (B_ * L_)        // 16384 rows
#define NB_  4
#define BSZ_ 128              // D_/NB_
#define MP_  8
#define LC_  64               // scan chunk length
#define NC_  (L_ / LC_)       // 32 chunks
#define BD_  (B_ * D_)        // 4096 channels

#define USE_ASYNC_LDS 1

typedef __attribute__((ext_vector_type(16))) __bf16 v16bf;
typedef __attribute__((ext_vector_type(8)))  __bf16 v8bf;
typedef __attribute__((ext_vector_type(8)))  float  v8f;

__device__ __forceinline__ float softplusf(float x) {
    return (x > 20.f) ? x : log1pf(expf(x));
}

// ---------------------------------------------------------------------------
// Split f32 -> bf16 hi + bf16 lo residual (for 3-term bf16 WMMA GEMM).
// ---------------------------------------------------------------------------
__global__ __launch_bounds__(256)
void split_bf16_kernel(const float* __restrict__ src,
                       __bf16* __restrict__ hi, __bf16* __restrict__ lo,
                       size_t n) {
    size_t i = (size_t)blockIdx.x * blockDim.x + threadIdx.x;
    size_t stride = (size_t)gridDim.x * blockDim.x;
    for (; i < n; i += stride) {
        float v = src[i];
        __bf16 h = (__bf16)v;
        hi[i] = h;
        lo[i] = (__bf16)(v - (float)h);
    }
}

// ---------------------------------------------------------------------------
// Pre-transpose + split weights: W (N,K) f32 -> Wt (K,N) bf16 hi/lo.
// ---------------------------------------------------------------------------
__global__ __launch_bounds__(256)
void split_wt_kernel(const float* __restrict__ W,     // (N,K)
                     __bf16* __restrict__ Wth,        // (K,N)
                     __bf16* __restrict__ Wtl,        // (K,N)
                     int N, int K) {
    size_t total = (size_t)N * K;
    size_t i = (size_t)blockIdx.x * blockDim.x + threadIdx.x;
    size_t stride = (size_t)gridDim.x * blockDim.x;
    for (; i < total; i += stride) {
        int k = (int)(i / N);
        int n = (int)(i % N);
        float w = W[(size_t)n * K + k];
        __bf16 h = (__bf16)w;
        Wth[i] = h;
        Wtl[i] = (__bf16)(w - (float)h);
    }
}

// ---------------------------------------------------------------------------
// GEMM: C[M,N] = A[M,K] * W^T + bias  (+C if ACC), A/W as bf16 hi/lo splits.
// Block tile 128x64, K-step 32, 8 waves (4x2), each wave = 2x2 WMMA tiles,
// 3 WMMA per tile per k-step (hi*hi + lo*hi + hi*lo).
// Branch-free double-buffered async-to-LDS pipeline: the steady-state loop
// always prefetches tile k+1 and waits asynccnt<=6 (tile k landed, tile k+1
// in flight); the last iteration is peeled with asynccnt<=0.
// Epilogue re-uses tile LDS as a 128x64 f32 buffer for coalesced stores.
// ---------------------------------------------------------------------------
#define GBM 128
#define GBN 64
#define GBK 32
#define XPAD 8
#define WPAD 8
#define XH_OFF 0
#define XL_OFF (XH_OFF + GBM * (GBK + XPAD) * 2)            // 10240
#define WH_OFF (XL_OFF + GBM * (GBK + XPAD) * 2)            // 20480
#define WL_OFF (WH_OFF + GBK * (GBN + WPAD) * 2)            // 25088
#define BUFB   (WL_OFF + GBK * (GBN + WPAD) * 2)            // 29696 per buffer
#define CS_STRIDE (GBN + 4)                                  // 68 floats
#define CS_BYTES (GBM * CS_STRIDE * 4)                       // 34816
#define SMEM_BYTES ((2 * BUFB) > CS_BYTES ? (2 * BUFB) : CS_BYTES)

template <int ACC>
__global__ __launch_bounds__(256)
void gemm_bf16x3_kernel(const __bf16* __restrict__ Ah,
                        const __bf16* __restrict__ Al,
                        const __bf16* __restrict__ Wth,   // (K,N)
                        const __bf16* __restrict__ Wtl,   // (K,N)
                        const float*  __restrict__ bias,
                        float* __restrict__ C,            // (M,N)
                        int M, int N, int K) {
    __shared__ __attribute__((aligned(16))) char smem[SMEM_BYTES];

    const int tid  = threadIdx.x;
    const int m0   = blockIdx.y * GBM;
    const int n0   = blockIdx.x * GBN;
    const int wave = tid >> 5;
    const int lane = tid & 31;
    const int wm   = wave >> 1;   // 0..3
    const int wn   = wave & 1;    // 0..1

    // staging coordinates (per thread, fixed)
    const int arow  = tid >> 1;          // 0..127  (A tile row)
    const int ahalf = (tid & 1) * 16;    // 0/16    (A tile col)
    const int wrow  = tid >> 3;          // 0..31   (W tile k-row)
    const int wseg  = (tid & 7) * 8;     // 0..56   (W tile n-col)

    // stage the k0 tile into buffer pbuf (0/1): 6 x 16B copies per thread
    auto stage = [&](int pbuf, int k0) {
        char* buf = smem + pbuf * BUFB;
        const __bf16* gxh = Ah  + (size_t)(m0 + arow) * K + k0 + ahalf;
        const __bf16* gxl = Al  + (size_t)(m0 + arow) * K + k0 + ahalf;
        const __bf16* gwh = Wth + (size_t)(k0 + wrow) * N + n0 + wseg;
        const __bf16* gwl = Wtl + (size_t)(k0 + wrow) * N + n0 + wseg;
        __bf16* lxh = (__bf16*)(buf + XH_OFF) + arow * (GBK + XPAD) + ahalf;
        __bf16* lxl = (__bf16*)(buf + XL_OFF) + arow * (GBK + XPAD) + ahalf;
        __bf16* lwh = (__bf16*)(buf + WH_OFF) + wrow * (GBN + WPAD) + wseg;
        __bf16* lwl = (__bf16*)(buf + WL_OFF) + wrow * (GBN + WPAD) + wseg;
#if USE_ASYNC_LDS
        unsigned oxh = (unsigned)(uintptr_t)lxh;
        unsigned oxl = (unsigned)(uintptr_t)lxl;
        unsigned owh = (unsigned)(uintptr_t)lwh;
        unsigned owl = (unsigned)(uintptr_t)lwl;
        asm volatile("global_load_async_to_lds_b128 %0, %1, off"
                     :: "v"(oxh), "v"(gxh) : "memory");
        asm volatile("global_load_async_to_lds_b128 %0, %1, off"
                     :: "v"(oxh + 16u), "v"(gxh + 8) : "memory");
        asm volatile("global_load_async_to_lds_b128 %0, %1, off"
                     :: "v"(oxl), "v"(gxl) : "memory");
        asm volatile("global_load_async_to_lds_b128 %0, %1, off"
                     :: "v"(oxl + 16u), "v"(gxl + 8) : "memory");
        asm volatile("global_load_async_to_lds_b128 %0, %1, off"
                     :: "v"(owh), "v"(gwh) : "memory");
        asm volatile("global_load_async_to_lds_b128 %0, %1, off"
                     :: "v"(owl), "v"(gwl) : "memory");
#else
        *(v8bf*)(lxh)     = *(const v8bf*)(gxh);
        *(v8bf*)(lxh + 8) = *(const v8bf*)(gxh + 8);
        *(v8bf*)(lxl)     = *(const v8bf*)(gxl);
        *(v8bf*)(lxl + 8) = *(const v8bf*)(gxl + 8);
        *(v8bf*)(lwh)     = *(const v8bf*)(gwh);
        *(v8bf*)(lwl)     = *(const v8bf*)(gwl);
#endif
    };

    v8f acc[2][2];
#pragma unroll
    for (int r = 0; r < 2; ++r)
#pragma unroll
        for (int c = 0; c < 2; ++c)
            acc[r][c] = (v8f){0.f, 0.f, 0.f, 0.f, 0.f, 0.f, 0.f, 0.f};

    // 12 WMMAs on the tile sitting in buffer `pbuf`
    auto compute = [&](int pbuf) {
        char* cb = smem + pbuf * BUFB;
        __bf16* Xh_s = (__bf16*)(cb + XH_OFF);
        __bf16* Xl_s = (__bf16*)(cb + XL_OFF);
        __bf16* Wh_s = (__bf16*)(cb + WH_OFF);
        __bf16* Wl_s = (__bf16*)(cb + WL_OFF);

        const int kh = (lane >> 4) * 8;     // K-half select per WMMA A layout
        const int mr = lane & 15;
        v16bf aH[2], aL[2], bH[2], bL[2];
#pragma unroll
        for (int r = 0; r < 2; ++r) {
            const __bf16* xr = Xh_s + (wm * 32 + r * 16 + mr) * (GBK + XPAD);
            v8bf p0 = *(const v8bf*)(xr + kh);
            v8bf p1 = *(const v8bf*)(xr + 16 + kh);
            aH[r] = __builtin_shufflevector(p0, p1, 0,1,2,3,4,5,6,7,8,9,10,11,12,13,14,15);
            const __bf16* xq = Xl_s + (wm * 32 + r * 16 + mr) * (GBK + XPAD);
            v8bf q0 = *(const v8bf*)(xq + kh);
            v8bf q1 = *(const v8bf*)(xq + 16 + kh);
            aL[r] = __builtin_shufflevector(q0, q1, 0,1,2,3,4,5,6,7,8,9,10,11,12,13,14,15);
        }
#pragma unroll
        for (int c = 0; c < 2; ++c) {
            int n = wn * 32 + c * 16;
            const __bf16* wr = Wh_s + n + mr;
            const __bf16* wq = Wl_s + n + mr;
#pragma unroll
            for (int s = 0; s < 16; ++s) {
                int kk = (s < 8) ? (kh + s) : (16 + kh + (s - 8));
                bH[c][s] = wr[kk * (GBN + WPAD)];
                bL[c][s] = wq[kk * (GBN + WPAD)];
            }
        }
#pragma unroll
        for (int r = 0; r < 2; ++r)
#pragma unroll
            for (int c = 0; c < 2; ++c) {
                acc[r][c] = __builtin_amdgcn_wmma_f32_16x16x32_bf16(
                    false, aH[r], false, bH[c], (short)0, acc[r][c], false, false);
                acc[r][c] = __builtin_amdgcn_wmma_f32_16x16x32_bf16(
                    false, aL[r], false, bH[c], (short)0, acc[r][c], false, false);
                acc[r][c] = __builtin_amdgcn_wmma_f32_16x16x32_bf16(
                    false, aH[r], false, bL[c], (short)0, acc[r][c], false, false);
            }
    };

    stage(0, 0);                       // prologue: prefetch first tile
    int parity = 0;

    // steady state: always prefetch next tile, branch-free body
    for (int k0 = 0; k0 < K - GBK; k0 += GBK) {
        stage(parity ^ 1, k0 + GBK);
#if USE_ASYNC_LDS
        asm volatile("s_wait_asynccnt 6" ::: "memory");   // tile k landed
#endif
        __syncthreads();
        compute(parity);
        __syncthreads();               // protect buffer prefetched next iter
        parity ^= 1;
    }
    // peeled last iteration: no prefetch
#if USE_ASYNC_LDS
    asm volatile("s_wait_asynccnt 0" ::: "memory");
#endif
    __syncthreads();
    compute(parity);
    __syncthreads();

    // ---- epilogue: stage C tile in LDS, then coalesced float4 stores ----
    float* Cs = (float*)smem;
#pragma unroll
    for (int r = 0; r < 2; ++r)
#pragma unroll
        for (int c = 0; c < 2; ++c) {
            int col = wn * 32 + c * 16 + (lane & 15);
            int rb  = wm * 32 + r * 16 + ((lane >> 4) * 8);
#pragma unroll
            for (int v = 0; v < 8; ++v)
                Cs[(rb + v) * CS_STRIDE + col] = acc[r][c][v];
        }
    __syncthreads();

    {
        const int row  = tid >> 1;
        const int half = (tid & 1) * 32;
        const float* csrow = Cs + row * CS_STRIDE + half;
        float* gout = C + (size_t)(m0 + row) * N + n0 + half;
        const float* bp = bias + n0 + half;
#pragma unroll
        for (int q = 0; q < 8; ++q) {
            float4 v = *(const float4*)(csrow + q * 4);
            float4 bv = *(const float4*)(bp + q * 4);
            v.x += bv.x; v.y += bv.y; v.z += bv.z; v.w += bv.w;
            if (ACC) {
                float4 o = *(const float4*)(gout + q * 4);
                v.x += o.x; v.y += o.y; v.z += o.z; v.w += o.w;
            }
            *(float4*)(gout + q * 4) = v;
        }
    }
}

// ---------------------------------------------------------------------------
// gate[m] = sigmoid(x[m,:] . w_gate + b_gate). One wave32 per row.
// ---------------------------------------------------------------------------
__global__ __launch_bounds__(256)
void gate_kernel(const float* __restrict__ x, const float* __restrict__ w,
                 const float* __restrict__ bg, float* __restrict__ gate) {
    int wave = threadIdx.x >> 5;
    int lane = threadIdx.x & 31;
    int row  = blockIdx.x * 8 + wave;
    const float* xp = x + (size_t)row * D_;
    float s = 0.f;
#pragma unroll 4
    for (int k = lane; k < D_; k += 32) s += xp[k] * w[k];
#pragma unroll
    for (int m = 16; m > 0; m >>= 1) s += __shfl_xor(s, m, 32);
    if (lane == 0) gate[row] = 1.f / (1.f + expf(-(s + bg[0])));
}

// ---------------------------------------------------------------------------
// Chunked complex scans: H[t] = a*H[t-1] + U[t], a = exp(lam*dt).
// ---------------------------------------------------------------------------
__device__ __forceinline__ void chan_decay(const float* om, const float* lg,
                                           float dta, int d, float& ar, float& ai) {
    float dec = expf(-expf(lg[d]) * dta);
    ar = dec * cosf(om[d] * dta);
    ai = dec * sinf(om[d] * dta);
}

__global__ __launch_bounds__(256)
void scan1a_kernel(const float* __restrict__ psi_re, const float* __restrict__ psi_im,
                   const float* __restrict__ bs_raw, const float* __restrict__ gate,
                   const float* __restrict__ omega,  const float* __restrict__ log_gamma,
                   const float* __restrict__ dtp,
                   float* __restrict__ Hp, float2* __restrict__ carry) {
    int gid = blockIdx.x * blockDim.x + threadIdx.x;   // c*BD + ch
    int ch = gid % BD_, c = gid / BD_;
    int b = ch / D_, d = ch % D_;
    float dta = fabsf(dtp[0]);
    float ar, ai; chan_decay(omega, log_gamma, dta, d, ar, ai);
    if (c == 0) {
        Hp[(size_t)b * L_ * (2 * D_) + d]      = 0.f;
        Hp[(size_t)b * L_ * (2 * D_) + D_ + d] = 0.f;
    }
    int t0 = c * LC_;
    float hr = 0.f, hi2 = 0.f;
    for (int i = 0; i < LC_; ++i) {
        int t = t0 + i;
        size_t row = (size_t)b * L_ + t;
        size_t base = row * D_ + d;
        float g  = gate[row];
        float bs = softplusf(bs_raw[base]);
        float ur = g * bs * psi_re[base];
        float ui = g * bs * psi_im[base];
        float nr = ar * hr - ai * hi2 + ur;
        float ni = ar * hi2 + ai * hr + ui;
        hr = nr; hi2 = ni;
        if (t + 1 < L_) {           // H_prev[t+1] = H[t]
            Hp[(row + 1) * (2 * D_) + d]      = hr;
            Hp[(row + 1) * (2 * D_) + D_ + d] = hi2;
        }
    }
    carry[gid] = make_float2(hr, hi2);
}

__global__ __launch_bounds__(256)
void scan_prefix_kernel(const float* __restrict__ omega, const float* __restrict__ log_gamma,
                        const float* __restrict__ dtp, float2* __restrict__ carry) {
    int ch = blockIdx.x * blockDim.x + threadIdx.x;    // 0..BD-1
    int d = ch % D_;
    float dta = fabsf(dtp[0]);
    float dec = expf(-expf(log_gamma[d]) * dta * (float)LC_);   // closed-form a^LC
    float ang = omega[d] * dta * (float)LC_;
    float a64r = dec * cosf(ang), a64i = dec * sinf(ang);
    float pr = 0.f, pi2 = 0.f;
    for (int c = 0; c < NC_; ++c) {
        float2 cr = carry[(size_t)c * BD_ + ch];
        carry[(size_t)c * BD_ + ch] = make_float2(pr, pi2);  // prefix for chunk c
        float npr = a64r * pr - a64i * pi2 + cr.x;
        float npi = a64r * pi2 + a64i * pr + cr.y;
        pr = npr; pi2 = npi;
    }
}

__global__ __launch_bounds__(256)
void scan1c_kernel(const float* __restrict__ omega, const float* __restrict__ log_gamma,
                   const float* __restrict__ dtp,
                   float* __restrict__ Hp, const float2* __restrict__ carry) {
    int gid = blockIdx.x * blockDim.x + threadIdx.x;
    int ch = gid % BD_, c = gid / BD_;
    int b = ch / D_, d = ch % D_;
    float2 p = carry[gid];
    if (p.x == 0.f && p.y == 0.f) return;
    float dta = fabsf(dtp[0]);
    float ar, ai; chan_decay(omega, log_gamma, dta, d, ar, ai);
    float wr = ar, wi = ai;
    int t0 = c * LC_;
    for (int i = 0; i < LC_; ++i) {
        int t = t0 + i;
        if (t + 1 < L_) {
            size_t ro = ((size_t)b * L_ + t + 1) * (2 * D_);
            Hp[ro + d]      += wr * p.x - wi * p.y;
            Hp[ro + D_ + d] += wr * p.y + wi * p.x;
        }
        float nwr = ar * wr - ai * wi;
        float nwi = ar * wi + ai * wr;
        wr = nwr; wi = nwi;
    }
}

__global__ __launch_bounds__(256)
void scan2a_kernel(const float* __restrict__ psi_re, const float* __restrict__ psi_im,
                   const float* __restrict__ bt_raw, const float* __restrict__ gate,
                   const float* __restrict__ omega,  const float* __restrict__ log_gamma,
                   const float* __restrict__ dtp,
                   float* __restrict__ H_re, float* __restrict__ H_im,
                   float2* __restrict__ carry) {
    int gid = blockIdx.x * blockDim.x + threadIdx.x;
    int ch = gid % BD_, c = gid / BD_;
    int b = ch / D_, d = ch % D_;
    float dta = fabsf(dtp[0]);
    float ar, ai; chan_decay(omega, log_gamma, dta, d, ar, ai);
    const float c16 = 1.f / 6.f, c46 = 4.f / 6.f;
    int t0 = c * LC_;
    float u1r = 0.f, u1i = 0.f, u2r = 0.f, u2i = 0.f;
#pragma unroll
    for (int j = 0; j < 2; ++j) {   // j=0 -> t0-1 (u1), j=1 -> t0-2 (u2)
        int t = t0 - 1 - j;
        float vr = 0.f, vi = 0.f;
        if (t >= 0) {
            size_t row = (size_t)b * L_ + t;
            size_t base = row * D_ + d;
            float g = gate[row] * softplusf(bt_raw[base]);
            vr = g * psi_re[base]; vi = g * psi_im[base];
        }
        if (j == 0) { u1r = vr; u1i = vi; } else { u2r = vr; u2i = vi; }
    }
    float hr = 0.f, hi2 = 0.f;
    for (int i = 0; i < LC_; ++i) {
        int t = t0 + i;
        size_t row = (size_t)b * L_ + t;
        size_t base = row * D_ + d;
        float g  = gate[row] * softplusf(bt_raw[base]);
        float ur = g * psi_re[base];
        float ui = g * psi_im[base];
        float usr = ur * c16 + u1r * c46 + u2r * c16;
        float usi = ui * c16 + u1i * c46 + u2i * c16;
        float nr = ar * hr - ai * hi2 + usr;
        float ni = ar * hi2 + ai * hr + usi;
        hr = nr; hi2 = ni;
        H_re[base] = hr;
        H_im[base] = hi2;
        u2r = u1r; u2i = u1i; u1r = ur; u1i = ui;
    }
    carry[gid] = make_float2(hr, hi2);
}

__global__ __launch_bounds__(256)
void scan2c_kernel(const float* __restrict__ omega, const float* __restrict__ log_gamma,
                   const float* __restrict__ dtp,
                   float* __restrict__ H_re, float* __restrict__ H_im,
                   const float2* __restrict__ carry) {
    int gid = blockIdx.x * blockDim.x + threadIdx.x;
    int ch = gid % BD_, c = gid / BD_;
    int b = ch / D_, d = ch % D_;
    float2 p = carry[gid];
    if (p.x == 0.f && p.y == 0.f) return;
    float dta = fabsf(dtp[0]);
    float ar, ai; chan_decay(omega, log_gamma, dta, d, ar, ai);
    float wr = ar, wi = ai;
    int t0 = c * LC_;
    for (int i = 0; i < LC_; ++i) {
        size_t base = ((size_t)b * L_ + t0 + i) * D_ + d;
        H_re[base] += wr * p.x - wi * p.y;
        H_im[base] += wr * p.y + wi * p.x;
        float nwr = ar * wr - ai * wi;
        float nwi = ar * wi + ai * wr;
        wr = nwr; wi = nwi;
    }
}

// ---------------------------------------------------------------------------
// Fused per-row tail: LayerNorm(re/im), band attention, 8x8 MIMO mix.
// ---------------------------------------------------------------------------
__device__ __forceinline__ float block_sum256(float v, float* red, int t) {
    red[t] = v; __syncthreads();
    for (int s = 128; s > 0; s >>= 1) {
        if (t < s) red[t] += red[t + s];
        __syncthreads();
    }
    float r = red[0]; __syncthreads();
    return r;
}

__global__ __launch_bounds__(256)
void row_tail_kernel(const float* __restrict__ H_re, const float* __restrict__ H_im,
                     const float* __restrict__ phi_re, const float* __restrict__ phi_im,
                     const int*   __restrict__ band_idx,
                     const float* __restrict__ tau, const float* __restrict__ beta,
                     const float* __restrict__ mimo_w, float* __restrict__ out) {
    __shared__ float hr[D_], hi2[D_], outr[D_], red[256];
    __shared__ float s_ar[NB_], s_ai[NB_], s_dn[NB_], s_c[NB_];

    const int t = threadIdx.x;
    const size_t base = (size_t)blockIdx.x * D_;

    float v0r = H_re[base + t], v1r = H_re[base + t + 256];
    float v0i = H_im[base + t], v1i = H_im[base + t + 256];

    float sr  = block_sum256(v0r + v1r, red, t);
    float sr2 = block_sum256(v0r * v0r + v1r * v1r, red, t);
    float si  = block_sum256(v0i + v1i, red, t);
    float si2 = block_sum256(v0i * v0i + v1i * v1i, red, t);
    const float invD = 1.f / (float)D_;
    float mr = sr * invD, mi = si * invD;
    float inr = rsqrtf(sr2 * invD - mr * mr + 1e-5f);
    float ini = rsqrtf(si2 * invD - mi * mi + 1e-5f);
    hr[t] = (v0r - mr) * inr;   hr[t + 256] = (v1r - mr) * inr;
    hi2[t] = (v0i - mi) * ini;  hi2[t + 256] = (v1i - mi) * ini;
    __syncthreads();

    const int k  = t >> 6;
    const int j0 = (t & 63) * 2;
    float par = 0.f, pai = 0.f, pd = 0.f;
#pragma unroll
    for (int jj = 0; jj < 2; ++jj) {
        int d = band_idx[k * BSZ_ + j0 + jj];
        float hbr = hr[d], hbi = hi2[d];
        float pr = phi_re[base + d], pi = phi_im[base + d];
        par += hbr * pr + hbi * pi;
        pai += hbr * pi - hbi * pr;
        pd  += hbr * hbr + hbi * hbi;
    }
    red[t] = par; __syncthreads();
    for (int s = 32; s > 0; s >>= 1) { if ((t & 63) < s) red[t] += red[t + s]; __syncthreads(); }
    if ((t & 63) == 0) s_ar[k] = red[t] * (1.f / (float)BSZ_);
    __syncthreads();
    red[t] = pai; __syncthreads();
    for (int s = 32; s > 0; s >>= 1) { if ((t & 63) < s) red[t] += red[t + s]; __syncthreads(); }
    if ((t & 63) == 0) s_ai[k] = red[t] * (1.f / (float)BSZ_);
    __syncthreads();
    red[t] = pd; __syncthreads();
    for (int s = 32; s > 0; s >>= 1) { if ((t & 63) < s) red[t] += red[t + s]; __syncthreads(); }
    if ((t & 63) == 0) s_dn[k] = red[t] * (1.f / (float)BSZ_);
    __syncthreads();

    if (t < NB_) {
        float tauc = fmaxf(tau[0], 1e-4f);
        float mags[NB_]; float mx = -1e30f;
#pragma unroll
        for (int q = 0; q < NB_; ++q) {
            mags[q] = sqrtf(s_ar[q] * s_ar[q] + s_ai[q] * s_ai[q]) / tauc;
            mx = fmaxf(mx, mags[q]);
        }
        float den = 0.f;
#pragma unroll
        for (int q = 0; q < NB_; ++q) den += expf(mags[q] - mx);
        s_c[t] = expf(mags[t] - mx) / den;
    }
    __syncthreads();

    {
        float afr = s_ar[k] * s_c[k] * (float)NB_ + beta[0] * s_dn[k];
        float afi = s_ai[k] * s_c[k] * (float)NB_;
#pragma unroll
        for (int jj = 0; jj < 2; ++jj) {
            int d = band_idx[k * BSZ_ + j0 + jj];
            outr[d] = hr[d] * afr - hi2[d] * afi;
        }
    }
    __syncthreads();

#pragma unroll
    for (int rep = 0; rep < 2; ++rep) {
        int idx = t + rep * 256;
        int g = idx >> 3, p = idx & 7;
        const float* wp = mimo_w + (g * MP_ + p) * MP_;
        float o = 0.f;
#pragma unroll
        for (int q = 0; q < MP_; ++q) o += wp[q] * outr[g * MP_ + q];
        out[base + idx] = o;
    }
}

// ---------------------------------------------------------------------------
// Host launcher
// ---------------------------------------------------------------------------
extern "C" void kernel_launch(void* const* d_in, const int* in_sizes, int n_in,
                              void* d_out, int out_size, void* d_ws, size_t ws_size,
                              hipStream_t stream) {
    (void)in_sizes; (void)n_in; (void)out_size; (void)ws_size;

    const float* x         = (const float*)d_in[0];
    const float* omega     = (const float*)d_in[1];
    const float* log_gamma = (const float*)d_in[2];
    const float* dt        = (const float*)d_in[3];
    const float* W_psi     = (const float*)d_in[4];
    const float* b_psi     = (const float*)d_in[5];
    const float* W_phi     = (const float*)d_in[6];
    const float* b_phi     = (const float*)d_in[7];
    const float* W_bs      = (const float*)d_in[8];
    const float* b_bs      = (const float*)d_in[9];
    const float* W_btx     = (const float*)d_in[10];
    const float* b_btx     = (const float*)d_in[11];
    const float* W_bth     = (const float*)d_in[12];
    const float* b_bth     = (const float*)d_in[13];
    const float* w_gate    = (const float*)d_in[14];
    const float* b_gate    = (const float*)d_in[15];
    const float* tau       = (const float*)d_in[16];
    const float* beta      = (const float*)d_in[17];
    const float* mimo_w    = (const float*)d_in[18];
    const int*   band_idx  = (const int*)d_in[19];
    float* out = (float*)d_out;

    char* ws = (char*)d_ws;
    size_t off = 0;
    auto take = [&](size_t bytes) -> char* {
        char* p = ws + off;
        off += (bytes + 255) & ~(size_t)255;
        return p;
    };
    const size_t MD = (size_t)M_ * D_;
    const size_t W55 = (size_t)512 * 512;      // 512x512 weight elems
    __bf16* xh     = (__bf16*)take(MD * 2);
    __bf16* xl     = (__bf16*)take(MD * 2);
    float*  psi_re = (float*)take(MD * 4);
    float*  psi_im = (float*)take(MD * 4);
    float*  phi_re = (float*)take(MD * 4);
    float*  phi_im = (float*)take(MD * 4);
    float*  bs_raw = (float*)take(MD * 4);
    float*  bt_raw = (float*)take(MD * 4);
    float*  gate   = (float*)take((size_t)M_ * 4);
    float*  Hp     = (float*)take(MD * 2 * 4);     // (M, 2D) [re|im]
    __bf16* Hph    = (__bf16*)take(MD * 2 * 2);
    __bf16* Hpl    = (__bf16*)take(MD * 2 * 2);
    float*  H_re   = (float*)take(MD * 4);
    float*  H_im   = (float*)take(MD * 4);
    float2* carry  = (float2*)take((size_t)BD_ * NC_ * 8);
    __bf16* wt_h[7]; __bf16* wt_l[7];
    for (int i = 0; i < 6; ++i) { wt_h[i] = (__bf16*)take(W55 * 2); wt_l[i] = (__bf16*)take(W55 * 2); }
    wt_h[6] = (__bf16*)take(W55 * 2 * 2);  wt_l[6] = (__bf16*)take(W55 * 2 * 2);  // bth: K=1024

    const dim3 blk(256);

    // 1) one-time splits
    split_bf16_kernel<<<4096, blk, 0, stream>>>(x, xh, xl, MD);
    split_wt_kernel<<<512, blk, 0, stream>>>(W_psi,       wt_h[0], wt_l[0], 512, 512);
    split_wt_kernel<<<512, blk, 0, stream>>>(W_psi + W55, wt_h[1], wt_l[1], 512, 512);
    split_wt_kernel<<<512, blk, 0, stream>>>(W_phi,       wt_h[2], wt_l[2], 512, 512);
    split_wt_kernel<<<512, blk, 0, stream>>>(W_phi + W55, wt_h[3], wt_l[3], 512, 512);
    split_wt_kernel<<<512, blk, 0, stream>>>(W_bs,        wt_h[4], wt_l[4], 512, 512);
    split_wt_kernel<<<512, blk, 0, stream>>>(W_btx,       wt_h[5], wt_l[5], 512, 512);
    split_wt_kernel<<<512, blk, 0, stream>>>(W_bth,       wt_h[6], wt_l[6], 512, 1024);

    // 2) gate
    gate_kernel<<<M_ / 8, blk, 0, stream>>>(x, w_gate, b_gate, gate);

    // 3) projections (WMMA GEMMs)
    const dim3 gg(512 / GBN, M_ / GBM);
    gemm_bf16x3_kernel<0><<<gg, blk, 0, stream>>>(xh, xl, wt_h[0], wt_l[0], b_psi,       psi_re, M_, 512, 512);
    gemm_bf16x3_kernel<0><<<gg, blk, 0, stream>>>(xh, xl, wt_h[1], wt_l[1], b_psi + 512, psi_im, M_, 512, 512);
    gemm_bf16x3_kernel<0><<<gg, blk, 0, stream>>>(xh, xl, wt_h[2], wt_l[2], b_phi,       phi_re, M_, 512, 512);
    gemm_bf16x3_kernel<0><<<gg, blk, 0, stream>>>(xh, xl, wt_h[3], wt_l[3], b_phi + 512, phi_im, M_, 512, 512);
    gemm_bf16x3_kernel<0><<<gg, blk, 0, stream>>>(xh, xl, wt_h[4], wt_l[4], b_bs,        bs_raw, M_, 512, 512);
    gemm_bf16x3_kernel<0><<<gg, blk, 0, stream>>>(xh, xl, wt_h[5], wt_l[5], b_btx,       bt_raw, M_, 512, 512);

    // 4) scout scan (chunked) -> shifted H_prev (M, 2D)
    scan1a_kernel<<<(BD_ * NC_) / 256, blk, 0, stream>>>(
        psi_re, psi_im, bs_raw, gate, omega, log_gamma, dt, Hp, carry);
    scan_prefix_kernel<<<BD_ / 256, blk, 0, stream>>>(omega, log_gamma, dt, carry);
    scan1c_kernel<<<(BD_ * NC_) / 256, blk, 0, stream>>>(omega, log_gamma, dt, Hp, carry);

    // 5) b_true = btx part + H_prev_flat @ W_bth^T + b_bth (accumulating GEMM)
    split_bf16_kernel<<<4096, blk, 0, stream>>>(Hp, Hph, Hpl, MD * 2);
    gemm_bf16x3_kernel<1><<<gg, blk, 0, stream>>>(Hph, Hpl, wt_h[6], wt_l[6], b_bth, bt_raw, M_, 512, 1024);

    // 6) true scan (chunked, with temporal smoothing) -> H
    scan2a_kernel<<<(BD_ * NC_) / 256, blk, 0, stream>>>(
        psi_re, psi_im, bt_raw, gate, omega, log_gamma, dt, H_re, H_im, carry);
    scan_prefix_kernel<<<BD_ / 256, blk, 0, stream>>>(omega, log_gamma, dt, carry);
    scan2c_kernel<<<(BD_ * NC_) / 256, blk, 0, stream>>>(omega, log_gamma, dt, H_re, H_im, carry);

    // 7) fused LN + band attention + MIMO
    row_tail_kernel<<<M_, blk, 0, stream>>>(
        H_re, H_im, phi_re, phi_im, band_idx, tau, beta, mimo_w, out);
}
